// TestModel_15530601742351
// MI455X (gfx1250) — compile-verified, matrix operand order checked by reading, not run on hardware
//
#include <hip/hip_runtime.h>
#include <hip/hip_bf16.h>

#define T_ROWS 8192
#define H_DIM  4096
#define FP8_MAX 448.0f
#define RMS_EPS 1e-6f

typedef __attribute__((ext_vector_type(16))) int   v16i;
typedef __attribute__((ext_vector_type(8)))  float v8f;
typedef int vsi4 __attribute__((vector_size(16)));   // matches builtin param type

#define AS(n) __attribute__((address_space(n)))

#if defined(__has_builtin)
# if __has_builtin(__builtin_amdgcn_global_load_async_to_lds_b128)
#  define HAVE_ASYNC_LDS 1
# endif
# if __has_builtin(__builtin_amdgcn_s_wait_asynccnt)
#  define HAVE_WAIT_ASYNC 1
# endif
#endif

// ---------------------------------------------------------------------------
// Async global -> LDS 16B copy (CDNA5 GLOBAL_LOAD_ASYNC_TO_LDS_B128),
// with synchronous fallback if the builtin is unavailable.
// ---------------------------------------------------------------------------
__device__ inline void cp16_async(const unsigned char* g, unsigned char* l) {
#if defined(HAVE_ASYNC_LDS)
    __builtin_amdgcn_global_load_async_to_lds_b128(
        (AS(1) vsi4*)(uintptr_t)g,
        (AS(3) vsi4*)(unsigned)(uintptr_t)l,   // flat LDS addr low 32 bits == LDS offset
        0, 0);
#else
    *(int4*)l = *(const int4*)g;
#endif
}

__device__ inline void wait_async_all() {
#if defined(HAVE_ASYNC_LDS)
# if defined(HAVE_WAIT_ASYNC)
    __builtin_amdgcn_s_wait_asynccnt(0);
# else
    asm volatile("s_wait_asynccnt 0x0" ::: "memory");
# endif
#endif
}

// ---------------------------------------------------------------------------
// f32 -> fp8 e4m3 conversion
// ---------------------------------------------------------------------------
__device__ inline unsigned e4m3_sw(float f) {
    unsigned u = __float_as_uint(f);
    unsigned s = (u >> 31) & 1u;
    float a = fabsf(f);
    a = fminf(a, FP8_MAX);
    if (a < 1e-9f) return s << 7;
    int e; float m = frexpf(a, &e);      // m in [0.5,1)
    int be = e - 1 + 7;
    unsigned mant;
    if (be <= 0) {                        // subnormal: step 2^-9
        mant = (unsigned)(a * 512.0f + 0.5f);
        if (mant > 7u) mant = 7u;
        return (s << 7) | mant;
    }
    mant = (unsigned)((m * 2.0f - 1.0f) * 8.0f + 0.5f);
    if (mant > 7u) { mant = 0u; be += 1; }
    if (be > 15) { be = 15; mant = 6u; }  // clamp to 448
    return (s << 7) | ((unsigned)be << 3) | mant;
}

__device__ inline unsigned pack4_e4m3(float a, float b, float c, float d) {
#if defined(__has_builtin) && __has_builtin(__builtin_amdgcn_cvt_pk_fp8_f32)
    int v = 0;
    v = __builtin_amdgcn_cvt_pk_fp8_f32(a, b, v, false);  // bytes 0,1
    v = __builtin_amdgcn_cvt_pk_fp8_f32(c, d, v, true);   // bytes 2,3
    return (unsigned)v;
#else
    return e4m3_sw(a) | (e4m3_sw(b) << 8) | (e4m3_sw(c) << 16) | (e4m3_sw(d) << 24);
#endif
}

// ---------------------------------------------------------------------------
// Weight quantize + transpose: w[k][n] (f32, row-major) -> wqT[n][k] (fp8)
// ---------------------------------------------------------------------------
__global__ void wquant_transpose(const float* __restrict__ w,
                                 unsigned char* __restrict__ wqT) {
    size_t idx = ((size_t)blockIdx.x * blockDim.x + threadIdx.x) * 4;
    int k = (int)(idx >> 12);       // / 4096
    int n = (int)(idx & 4095);
    float4 f = *(const float4*)(w + idx);
    unsigned p = pack4_e4m3(f.x, f.y, f.z, f.w);
    wqT[(size_t)(n + 0) * H_DIM + k] = (unsigned char)(p & 0xff);
    wqT[(size_t)(n + 1) * H_DIM + k] = (unsigned char)((p >> 8) & 0xff);
    wqT[(size_t)(n + 2) * H_DIM + k] = (unsigned char)((p >> 16) & 0xff);
    wqT[(size_t)(n + 3) * H_DIM + k] = (unsigned char)((p >> 24) & 0xff);
}

// ---------------------------------------------------------------------------
// Row reduction helper (wave32 shuffle + cross-wave LDS)
// ---------------------------------------------------------------------------
__device__ inline float block_sum256(float v) {
    #pragma unroll
    for (int off = 16; off > 0; off >>= 1)
        v += __shfl_xor(v, off, 32);
    __shared__ float red[8];
    int wave = threadIdx.x >> 5, lane = threadIdx.x & 31;
    if (lane == 0) red[wave] = v;
    __syncthreads();
    float t = 0.0f;
    #pragma unroll
    for (int i = 0; i < 8; ++i) t += red[i];
    return t;
}

// ---------------------------------------------------------------------------
// Stage 0: relu + rmsnorm + quantize; also writes resid = relu(x)
// ---------------------------------------------------------------------------
__global__ void relu_rms_quant(const float* __restrict__ x,
                               const float* __restrict__ g,
                               const float* __restrict__ s_ptr,
                               float* __restrict__ resid,
                               unsigned char* __restrict__ yq) {
    const int row = blockIdx.x;
    const int t   = threadIdx.x;
    const float4* xr = (const float4*)(x + (size_t)row * H_DIM);
    float4*       rr = (float4*)(resid + (size_t)row * H_DIM);
    float4 v[4];
    float ss = 0.0f;
    #pragma unroll
    for (int c = 0; c < 4; ++c) {
        float4 f = xr[c * 256 + t];
        f.x = fmaxf(f.x, 0.0f); f.y = fmaxf(f.y, 0.0f);
        f.z = fmaxf(f.z, 0.0f); f.w = fmaxf(f.w, 0.0f);
        v[c] = f;
        rr[c * 256 + t] = f;
        ss += f.x * f.x + f.y * f.y + f.z * f.z + f.w * f.w;
    }
    float total = block_sum256(ss);
    float rstd  = rsqrtf(total * (1.0f / H_DIM) + RMS_EPS);
    float inv_s = 1.0f / s_ptr[0];
    const float4* gr = (const float4*)g;
    unsigned* qr = (unsigned*)(yq + (size_t)row * H_DIM);
    #pragma unroll
    for (int c = 0; c < 4; ++c) {
        float4 gv = gr[c * 256 + t];
        float4 f  = v[c];
        float a0 = fminf(fmaxf(f.x * rstd * gv.x * inv_s, -FP8_MAX), FP8_MAX);
        float a1 = fminf(fmaxf(f.y * rstd * gv.y * inv_s, -FP8_MAX), FP8_MAX);
        float a2 = fminf(fmaxf(f.z * rstd * gv.z * inv_s, -FP8_MAX), FP8_MAX);
        float a3 = fminf(fmaxf(f.w * rstd * gv.w * inv_s, -FP8_MAX), FP8_MAX);
        qr[c * 256 + t] = pack4_e4m3(a0, a1, a2, a3);
    }
}

// ---------------------------------------------------------------------------
// Mid stages: rmsnorm(resid) + quantize
// ---------------------------------------------------------------------------
__global__ void rms_quant(const float* __restrict__ resid,
                          const float* __restrict__ g,
                          const float* __restrict__ s_ptr,
                          unsigned char* __restrict__ yq) {
    const int row = blockIdx.x;
    const int t   = threadIdx.x;
    const float4* rr = (const float4*)(resid + (size_t)row * H_DIM);
    float4 v[4];
    float ss = 0.0f;
    #pragma unroll
    for (int c = 0; c < 4; ++c) {
        float4 f = rr[c * 256 + t];
        v[c] = f;
        ss += f.x * f.x + f.y * f.y + f.z * f.z + f.w * f.w;
    }
    float total = block_sum256(ss);
    float rstd  = rsqrtf(total * (1.0f / H_DIM) + RMS_EPS);
    float inv_s = 1.0f / s_ptr[0];
    const float4* gr = (const float4*)g;
    unsigned* qr = (unsigned*)(yq + (size_t)row * H_DIM);
    #pragma unroll
    for (int c = 0; c < 4; ++c) {
        float4 gv = gr[c * 256 + t];
        float4 f  = v[c];
        float a0 = fminf(fmaxf(f.x * rstd * gv.x * inv_s, -FP8_MAX), FP8_MAX);
        float a1 = fminf(fmaxf(f.y * rstd * gv.y * inv_s, -FP8_MAX), FP8_MAX);
        float a2 = fminf(fmaxf(f.z * rstd * gv.z * inv_s, -FP8_MAX), FP8_MAX);
        float a3 = fminf(fmaxf(f.w * rstd * gv.w * inv_s, -FP8_MAX), FP8_MAX);
        qr[c * 256 + t] = pack4_e4m3(a0, a1, a2, a3);
    }
}

// ---------------------------------------------------------------------------
// Final stage: y4 = rmsnorm(resid, g3) -> f32 output
// ---------------------------------------------------------------------------
__global__ void rms_out(const float* __restrict__ resid,
                        const float* __restrict__ g,
                        float* __restrict__ out) {
    const int row = blockIdx.x;
    const int t   = threadIdx.x;
    const float4* rr = (const float4*)(resid + (size_t)row * H_DIM);
    float4 v[4];
    float ss = 0.0f;
    #pragma unroll
    for (int c = 0; c < 4; ++c) {
        float4 f = rr[c * 256 + t];
        v[c] = f;
        ss += f.x * f.x + f.y * f.y + f.z * f.z + f.w * f.w;
    }
    float total = block_sum256(ss);
    float rstd  = rsqrtf(total * (1.0f / H_DIM) + RMS_EPS);
    const float4* gr = (const float4*)g;
    float4* orow = (float4*)(out + (size_t)row * H_DIM);
    #pragma unroll
    for (int c = 0; c < 4; ++c) {
        float4 gv = gr[c * 256 + t];
        float4 f  = v[c];
        float4 o;
        o.x = f.x * rstd * gv.x; o.y = f.y * rstd * gv.y;
        o.z = f.z * rstd * gv.z; o.w = f.w * rstd * gv.w;
        orow[c * 256 + t] = o;
    }
}

// ---------------------------------------------------------------------------
// fp8 GEMM + residual: resid[T,H] += (s*ws) * (Aq[T,H] @ BqT[N,K]^T)
// 128x128x128 tile, 8 waves (2Mx4N), per-wave 4x2 WMMA 16x16x128 tiles.
// Double-buffered LDS fed by GLOBAL_LOAD_ASYNC_TO_LDS_B128 (ASYNCcnt).
// ---------------------------------------------------------------------------
#define LDSK 144   // 128B K-slab + 16B pad (bank-conflict stagger)
#define NKT  (H_DIM / 128)

__device__ inline v16i load_frag64(const unsigned char* p) {
    int4 x0 = *(const int4*)(p);
    int4 x1 = *(const int4*)(p + 16);
    int4 x2 = *(const int4*)(p + 32);
    int4 x3 = *(const int4*)(p + 48);
    v16i r;
    r[0]=x0.x;  r[1]=x0.y;  r[2]=x0.z;  r[3]=x0.w;
    r[4]=x1.x;  r[5]=x1.y;  r[6]=x1.z;  r[7]=x1.w;
    r[8]=x2.x;  r[9]=x2.y;  r[10]=x2.z; r[11]=x2.w;
    r[12]=x3.x; r[13]=x3.y; r[14]=x3.z; r[15]=x3.w;
    return r;
}

__global__ __launch_bounds__(256, 2)
void gemm_resid(const unsigned char* __restrict__ Aq,   // [T, K] fp8
                const unsigned char* __restrict__ BqT,  // [N, K] fp8 (pre-transposed weight)
                const float* __restrict__ s_ptr,
                const float* __restrict__ ws_ptr,
                float* __restrict__ resid) {            // [T, N] f32, updated in place
    __shared__ unsigned char lA[2][128 * LDSK];
    __shared__ unsigned char lB[2][128 * LDSK];

    const int tid  = threadIdx.x;
    const int wave = tid >> 5;
    const int lane = tid & 31;
    const int wm   = wave & 1;        // wave grid 2 (M) x 4 (N)
    const int wn   = wave >> 1;
    const int m0   = blockIdx.x * 128;
    const int n0   = blockIdx.y * 128;
    const float alpha = s_ptr[0] * ws_ptr[0];

    // global staging coords: thread -> (row 0..31 [+32*c], 16B chunk 0..7)
    const int lrow = tid >> 3;
    const int lcol = (tid & 7) * 16;
    const unsigned char* gA = Aq  + (size_t)(m0 + lrow) * H_DIM + lcol;
    const unsigned char* gB = BqT + (size_t)(n0 + lrow) * H_DIM + lcol;
    const size_t rstep = (size_t)32 * H_DIM;   // +32 rows

    v8f acc[4][2];
    #pragma unroll
    for (int mi = 0; mi < 4; ++mi)
        #pragma unroll
        for (int ni = 0; ni < 2; ++ni)
            acc[mi][ni] = (v8f){};

    // prologue: tile 0 -> buffer 0
    #pragma unroll
    for (int c = 0; c < 4; ++c) {
        cp16_async(gA + c * rstep, &lA[0][(lrow + c * 32) * LDSK + lcol]);
        cp16_async(gB + c * rstep, &lB[0][(lrow + c * 32) * LDSK + lcol]);
    }

    for (int kt = 0; kt < NKT; ++kt) {
        const int cur = kt & 1;
        wait_async_all();     // my tile-kt async writes have landed in LDS
        __syncthreads();      // everyone's landed; prev reads of buf^1 done

        if (kt + 1 < NKT) {   // stream tile kt+1 into the other buffer
            const int koff = (kt + 1) * 128;
            #pragma unroll
            for (int c = 0; c < 4; ++c) {
                cp16_async(gA + koff + c * rstep,
                           &lA[cur ^ 1][(lrow + c * 32) * LDSK + lcol]);
                cp16_async(gB + koff + c * rstep,
                           &lB[cur ^ 1][(lrow + c * 32) * LDSK + lcol]);
            }
        }

        // fragment fetch: lane L -> row (L&15), K-half (L>>4)*64 bytes
        const int fr = lane & 15;
        const int fo = (lane >> 4) * 64;
        v16i a[4], b[2];
        #pragma unroll
        for (int mi = 0; mi < 4; ++mi)
            a[mi] = load_frag64(&lA[cur][(wm * 64 + mi * 16 + fr) * LDSK + fo]);
        #pragma unroll
        for (int ni = 0; ni < 2; ++ni)
            b[ni] = load_frag64(&lB[cur][(wn * 32 + ni * 16 + fr) * LDSK + fo]);

        #pragma unroll
        for (int mi = 0; mi < 4; ++mi)
            #pragma unroll
            for (int ni = 0; ni < 2; ++ni)
                acc[mi][ni] = __builtin_amdgcn_wmma_f32_16x16x128_fp8_fp8(
                    a[mi], b[ni], (short)0, acc[mi][ni],
                    /*reuse_a=*/false, /*reuse_b=*/false);
    }

    // epilogue: C/D layout — VGPR r, lane L -> (M = r + (L>>4)*8, N = L&15)
    const int cn  = lane & 15;
    const int cm8 = (lane >> 4) * 8;
    #pragma unroll
    for (int mi = 0; mi < 4; ++mi) {
        #pragma unroll
        for (int ni = 0; ni < 2; ++ni) {
            int col = n0 + wn * 32 + ni * 16 + cn;
            #pragma unroll
            for (int r = 0; r < 8; ++r) {
                int row = m0 + wm * 64 + mi * 16 + cm8 + r;
                size_t idx = (size_t)row * H_DIM + col;
                resid[idx] += alpha * acc[mi][ni][r];
            }
        }
    }
}

// ---------------------------------------------------------------------------
// Launcher
// ---------------------------------------------------------------------------
extern "C" void kernel_launch(void* const* d_in, const int* in_sizes, int n_in,
                              void* d_out, int out_size, void* d_ws, size_t ws_size,
                              hipStream_t stream) {
    const float* x  = (const float*)d_in[0];
    const float* g0 = (const float*)d_in[1];
    const float* g1 = (const float*)d_in[2];
    const float* g2 = (const float*)d_in[3];
    const float* g3 = (const float*)d_in[4];
    const float* w[3]   = { (const float*)d_in[5],  (const float*)d_in[8],  (const float*)d_in[11] };
    const float* wsp[3] = { (const float*)d_in[6],  (const float*)d_in[9],  (const float*)d_in[12] };
    const float* sp[3]  = { (const float*)d_in[7],  (const float*)d_in[10], (const float*)d_in[13] };

    // workspace carve (256B aligned)
    uintptr_t base = ((uintptr_t)d_ws + 255) & ~(uintptr_t)255;
    const size_t WQ  = (size_t)H_DIM * H_DIM;          // 16 MB fp8 each
    const size_t YQ  = (size_t)T_ROWS * H_DIM;         // 32 MB fp8
    unsigned char* wq0 = (unsigned char*)base;
    unsigned char* wq1 = wq0 + WQ;
    unsigned char* wq2 = wq1 + WQ;
    unsigned char* yq  = wq2 + WQ;
    float*         resid = (float*)((uintptr_t)(yq + YQ + 255) & ~(uintptr_t)255);
    unsigned char* wq[3] = { wq0, wq1, wq2 };

    // 1) one-time weight quantize + transpose
    const int wq_blocks = (int)(WQ / 4 / 256);
    for (int i = 0; i < 3; ++i)
        wquant_transpose<<<wq_blocks, 256, 0, stream>>>(w[i], wq[i]);

    dim3 gg(T_ROWS / 128, H_DIM / 128);

    // 2) stage 0: relu + rmsnorm(g0) + quant(s0), resid = relu(x)
    relu_rms_quant<<<T_ROWS, 256, 0, stream>>>(x, g0, sp[0], resid, yq);
    gemm_resid<<<gg, 256, 0, stream>>>(yq, wq[0], sp[0], wsp[0], resid);

    // 3) stage 1
    rms_quant<<<T_ROWS, 256, 0, stream>>>(resid, g1, sp[1], yq);
    gemm_resid<<<gg, 256, 0, stream>>>(yq, wq[1], sp[1], wsp[1], resid);

    // 4) stage 2
    rms_quant<<<T_ROWS, 256, 0, stream>>>(resid, g2, sp[2], yq);
    gemm_resid<<<gg, 256, 0, stream>>>(yq, wq[2], sp[2], wsp[2], resid);

    // 5) final rmsnorm(g3) -> output
    rms_out<<<T_ROWS, 256, 0, stream>>>(resid, g3, (float*)d_out);
}
